// FusionNet_17360257810492
// MI455X (gfx1250) — matrix operand order repaired and checked by reading, not compiled, not run on hardware
//
#include <hip/hip_runtime.h>

// ============================================================================
// MI455X (gfx1250) fused dual-ResNet + encoder + kmeans reference.
// - Dense math on v_wmma_f32_16x16x32_f16 (wave32, 8 waves/block, 64x64x32).
// - B (weight) tiles staged LDS-side by the Tensor Data Mover
//   (tensor_load_to_lds + s_wait_tensorcnt), weights pre-converted to f16.
//   Row-major 64x32 f16 tile => each lane's B fragment is 32 contiguous bytes
//   (lane = N, K = elem + 16*hi) -> 2x ds_load_b128, no reorder needed.
// - A tiles (fp32 activations, im2col) converted in registers and stored in
//   fragment order (2x ds_load_b128 per fragment).
// - All index decodes unsigned (avoids signed-div sign-fixup VALU chains).
//
// Roofline: ~300 GFLOP vs ~2 GB HBM -> f16 WMMA w/ f32 accumulate.
//
// d_in layout assumption (dict order; jax sorted-key flatten inside params):
//   0 x1, 1 x2, 2 label, 3 cls_b, 4 cls_w, 5..40 enc(3x12),
//   41..102 r1, 103..164 r2 (see offsets below).
// ============================================================================

typedef __attribute__((ext_vector_type(16))) _Float16 v16h;
typedef __attribute__((ext_vector_type(8)))  _Float16 h8;
typedef __attribute__((ext_vector_type(8)))  float    v8f;
typedef __attribute__((ext_vector_type(4)))  float    f4v;
typedef __attribute__((ext_vector_type(4)))  unsigned v4u_t;
typedef __attribute__((ext_vector_type(8)))  int      v8i_t;
typedef __attribute__((ext_vector_type(4)))  int      v4i_t;

#define BM 64
#define BN 64
#define BKK 32

#if __has_builtin(__builtin_amdgcn_tensor_load_to_lds) && \
    __has_builtin(__builtin_amdgcn_s_wait_tensorcnt)
#define USE_TDM 1
#else
#define USE_TDM 0
#endif

__device__ __forceinline__ h8 cvt8(f4v a, f4v b) {
  h8 r;
  r[0] = (_Float16)a[0]; r[1] = (_Float16)a[1];
  r[2] = (_Float16)a[2]; r[3] = (_Float16)a[3];
  r[4] = (_Float16)b[0]; r[5] = (_Float16)b[1];
  r[6] = (_Float16)b[2]; r[7] = (_Float16)b[3];
  return r;
}

#if USE_TDM
// TDM: copy a 64(N) x 32(K) f16 tile of W[N][K] starting at (n0,k0) into LDS
// (row-major, 64B rows). D# per CDNA5 ISA.
__device__ __forceinline__ void tdm_load_tile_b(
    const _Float16* Wh, unsigned ldsByte, int N, int K, int n0, int k0)
{
  unsigned long long ga =
      (unsigned long long)(uintptr_t)(Wh + (size_t)n0 * K + k0);
  v4u_t g0;
  g0[0] = 1u;                                         // count=1 (valid user D#)
  g0[1] = ldsByte;                                    // lds_addr
  g0[2] = (unsigned)(ga & 0xffffffffu);               // global_addr[31:0]
  g0[3] = (unsigned)((ga >> 32) & 0x1ffffffu) | (2u << 30);  // [56:32] | type=2
  v8i_t g1;
  g1[0] = (int)(1u << 16);                            // data_size=1 -> 2 bytes
  g1[1] = (int)(((unsigned)K & 0xffffu) << 16);       // tensor_dim0[15:0]
  g1[2] = (int)((((unsigned)K >> 16) & 0xffffu) |
                (((unsigned)N & 0xffffu) << 16));     // dim0 hi | dim1 lo
  g1[3] = (int)((((unsigned)N >> 16) & 0xffffu) |
                (32u << 16));                         // dim1 hi | tile_dim0=32
  g1[4] = 64;                                         // tile_dim1=64, tile_dim2=0
  g1[5] = K;                                          // tensor_dim0_stride[31:0]
  g1[6] = 0;                                          // stride hi | dim1_stride lo
  g1[7] = 0;
  v4i_t z4 = {0, 0, 0, 0};
#if defined(__clang_major__) && __clang_major__ >= 23
  v8i_t z8 = {0, 0, 0, 0, 0, 0, 0, 0};
  __builtin_amdgcn_tensor_load_to_lds(g0, g1, z4, z4, z8, 0);
#else
  __builtin_amdgcn_tensor_load_to_lds(g0, g1, z4, z4, 0);
#endif
}
#endif

// ---------------------------------------------------------------------------
// Implicit-GEMM convolution: M = B*OH*OW, K = Cin*KH*KW, N = Cout.
// Weights pre-converted to f16 (Wh). Writes raw conv output (+optional bias).
// ---------------------------------------------------------------------------
__global__ __launch_bounds__(256) void conv_igemm_f16_wmma(
    const float* __restrict__ X, const _Float16* __restrict__ Wh,
    const float* __restrict__ bias, float* __restrict__ Y,
    int Bn, int Cin, int H, int W, int Cout, int OH, int OW,
    int KH, int KW, int stride, int pad)
{
  __shared__ _Float16 AsF[4 * 512];    // fragment-order A tiles
  __shared__ _Float16 Bt[64 * 32];     // raw row-major B tile (TDM target)
  const int M = Bn * OH * OW;
  const int K = Cin * KH * KW;
  const int tid  = threadIdx.x;
  const int lane = tid & 31;
  const int hi   = lane >> 4;
  const int wave = tid >> 5;
  const int wm   = wave & 3;
  const int wn0  = (wave >> 2) * 2;
  const int m0 = blockIdx.x * BM;
  const int n0 = blockIdx.y * BN;

  const int mm   = tid & 63;
  const int kseg = tid >> 6;
  const int aoff = (mm >> 4) * 512 + ((mm & 15) + 16 * (kseg >> 1)) * 16 + (kseg & 1) * 8;

  // hoisted output-pixel decode (unsigned: no sign-fixup chains)
  const unsigned mg = (unsigned)(m0 + mm);
  const bool mok = (mg < (unsigned)M);
  unsigned bidx = 0, oh = 0, ow = 0;
  if (mok) {
    unsigned ohw = (unsigned)(OH * OW);
    bidx = mg / ohw;
    unsigned pr = mg % ohw;
    oh = pr / (unsigned)OW; ow = pr % (unsigned)OW;
  }
  const int ng = n0 + mm;

  v8f acc0 = {}; v8f acc1 = {};

  for (int k0 = 0; k0 < K; k0 += BKK) {
    const int kg = k0 + kseg * 8;
#if USE_TDM
    if (tid < 32) tdm_load_tile_b(Wh, (unsigned)(uintptr_t)&Bt[0], Cout, K, n0, k0);
#endif
    // ---- A: im2col gather, 8 consecutive k for one output pixel ----
    h8 av = {};
    if (mok && kg < K) {
      unsigned khw = (unsigned)(KH * KW);
      unsigned cin = (unsigned)kg / khw;
      unsigned kr  = (unsigned)kg % khw;
      unsigned kh = kr / (unsigned)KW, kw = kr % (unsigned)KW;
#pragma unroll
      for (int e = 0; e < 8; ++e) {
        if (kg + e < K) {
          int ih = (int)(oh * stride) - pad + (int)kh;
          int iw = (int)(ow * stride) - pad + (int)kw;
          if (ih >= 0 && ih < H && iw >= 0 && iw < W)
            av[e] = (_Float16)X[(((int)bidx * Cin + (int)cin) * H + ih) * W + iw];
        }
        if (++kw == (unsigned)KW) { kw = 0; if (++kh == (unsigned)KH) { kh = 0; ++cin; } }
      }
    }
    *(h8*)&AsF[aoff] = av;
#if !USE_TDM
    {
      h8 bv = {};
      if (ng < Cout) {
        if ((K & 7) == 0) {
          bv = *(const h8*)(Wh + (size_t)ng * K + kg);
        } else {
#pragma unroll
          for (int e = 0; e < 8; ++e)
            if (kg + e < K) bv[e] = Wh[(size_t)ng * K + kg + e];
        }
      }
      Bt[mm * 32 + kseg * 8 + 0] = bv[0]; // keep raw-tile layout
      *(h8*)&Bt[mm * 32 + kseg * 8] = bv;
    }
#endif
#if USE_TDM
    if (tid < 32) __builtin_amdgcn_s_wait_tensorcnt(0);
#endif
    __syncthreads();

    v16h fa  = *(const v16h*)&AsF[wm * 512 + lane * 16];
    const int bl = (lane & 15) * 32 + hi * 16;
    v16h fb0 = *(const v16h*)&Bt[(wn0 * 16) * 32 + bl];
    v16h fb1 = *(const v16h*)&Bt[((wn0 + 1) * 16) * 32 + bl];
    acc0 = __builtin_amdgcn_wmma_f32_16x16x32_f16(false, fa, false, fb0,
                                                  (short)0, acc0, false, false);
    acc1 = __builtin_amdgcn_wmma_f32_16x16x32_f16(false, fa, false, fb1,
                                                  (short)0, acc1, false, false);
    __syncthreads();
  }

  // ---- epilogue: D layout M = r + 8*hi, N = lane&15; scatter NCHW ----
#pragma unroll
  for (int r = 0; r < 8; ++r) {
    unsigned mgo = (unsigned)(m0 + wm * 16 + r + 8 * hi);
    if (mgo >= (unsigned)M) continue;
    unsigned ohw = (unsigned)(OH * OW);
    unsigned bo = mgo / ohw;
    unsigned pr = mgo % ohw;
    unsigned oho = pr / (unsigned)OW, owo = pr % (unsigned)OW;
    int ng0 = n0 + wn0 * 16 + (lane & 15);
    int ng1 = ng0 + 16;
    if (ng0 < Cout) {
      float v = acc0[r] + (bias ? bias[ng0] : 0.f);
      Y[(((int)bo * Cout + ng0) * OH + (int)oho) * OW + (int)owo] = v;
    }
    if (ng1 < Cout) {
      float v = acc1[r] + (bias ? bias[ng1] : 0.f);
      Y[(((int)bo * Cout + ng1) * OH + (int)oho) * OW + (int)owo] = v;
    }
  }
}

// ---------------------------------------------------------------------------
// Dense GEMM: C = act(A[M,K] * Wh[N,K]^T + bias). K multiple of 32 here.
// ---------------------------------------------------------------------------
__global__ __launch_bounds__(256) void gemm_f16_wmma(
    const float* __restrict__ A, const _Float16* __restrict__ Wh,
    const float* __restrict__ bias, float* __restrict__ C,
    int M, int N, int K, int relu)
{
  __shared__ _Float16 AsF[4 * 512];
  __shared__ _Float16 Bt[64 * 32];
  const int tid  = threadIdx.x;
  const int lane = tid & 31;
  const int hi   = lane >> 4;
  const int wave = tid >> 5;
  const int wm   = wave & 3;
  const int wn0  = (wave >> 2) * 2;
  const int m0 = blockIdx.x * BM;
  const int n0 = blockIdx.y * BN;

  const int mm   = tid & 63;
  const int kseg = tid >> 6;
  const int aoff = (mm >> 4) * 512 + ((mm & 15) + 16 * (kseg >> 1)) * 16 + (kseg & 1) * 8;
  const int mg = m0 + mm;
  const int ng = n0 + mm;

  v8f acc0 = {}; v8f acc1 = {};

  for (int k0 = 0; k0 < K; k0 += BKK) {
    const int kg = k0 + kseg * 8;
#if USE_TDM
    if (tid < 32) tdm_load_tile_b(Wh, (unsigned)(uintptr_t)&Bt[0], N, K, n0, k0);
#endif
    __builtin_prefetch(&A[(size_t)mg * K + kg + BKK], 0, 3);
    h8 av = {};
    if (mg < M) {
      const float* ap = A + (size_t)mg * K + kg;
      av = cvt8(*(const f4v*)ap, *(const f4v*)(ap + 4));
    }
    *(h8*)&AsF[aoff] = av;
#if !USE_TDM
    {
      h8 bv = {};
      if (ng < N) bv = *(const h8*)(Wh + (size_t)ng * K + kg);
      *(h8*)&Bt[mm * 32 + kseg * 8] = bv;
    }
#endif
#if USE_TDM
    if (tid < 32) __builtin_amdgcn_s_wait_tensorcnt(0);
#endif
    __syncthreads();

    v16h fa  = *(const v16h*)&AsF[wm * 512 + lane * 16];
    const int bl = (lane & 15) * 32 + hi * 16;
    v16h fb0 = *(const v16h*)&Bt[(wn0 * 16) * 32 + bl];
    v16h fb1 = *(const v16h*)&Bt[((wn0 + 1) * 16) * 32 + bl];
    acc0 = __builtin_amdgcn_wmma_f32_16x16x32_f16(false, fa, false, fb0,
                                                  (short)0, acc0, false, false);
    acc1 = __builtin_amdgcn_wmma_f32_16x16x32_f16(false, fa, false, fb1,
                                                  (short)0, acc1, false, false);
    __syncthreads();
  }

#pragma unroll
  for (int r = 0; r < 8; ++r) {
    int mgo = m0 + wm * 16 + r + 8 * hi;
    if (mgo >= M) continue;
    int ng0 = n0 + wn0 * 16 + (lane & 15);
    int ng1 = ng0 + 16;
    if (ng0 < N) {
      float v = acc0[r] + (bias ? bias[ng0] : 0.f);
      if (relu) v = fmaxf(v, 0.f);
      C[(size_t)mgo * N + ng0] = v;
    }
    if (ng1 < N) {
      float v = acc1[r] + (bias ? bias[ng1] : 0.f);
      if (relu) v = fmaxf(v, 0.f);
      C[(size_t)mgo * N + ng1] = v;
    }
  }
}

// ---------------------------------------------------------------------------
__global__ void f32_to_f16_kernel(const float* __restrict__ X,
                                  _Float16* __restrict__ Y, int n) {
  int i = blockIdx.x * blockDim.x + threadIdx.x;
  if (i < n) Y[i] = (_Float16)X[i];
}

__global__ __launch_bounds__(256) void bn_stats_kernel(
    const float* __restrict__ X, float* __restrict__ mean,
    float* __restrict__ invstd, int Bn, int C, int HW)
{
  __shared__ float s1[256], s2[256];
  int c = blockIdx.x, t = threadIdx.x;
  float a1 = 0.f, a2 = 0.f;
  unsigned Np = (unsigned)(Bn * HW);
  for (unsigned i = t; i < Np; i += 256) {
    unsigned b = i / (unsigned)HW, r = i % (unsigned)HW;
    float v = X[((size_t)(b * C + c)) * HW + r];
    a1 += v; a2 += v * v;
  }
  s1[t] = a1; s2[t] = a2; __syncthreads();
  for (int o = 128; o > 0; o >>= 1) {
    if (t < o) { s1[t] += s1[t + o]; s2[t] += s2[t + o]; }
    __syncthreads();
  }
  if (t == 0) {
    float m = s1[0] / Np;
    float var = s2[0] / Np - m * m;
    mean[c] = m; invstd[c] = rsqrtf(var + 1e-5f);
  }
}

__global__ void bn_apply_kernel(
    float* __restrict__ X, const float* __restrict__ R,
    const float* __restrict__ mean, const float* __restrict__ invstd,
    const float* __restrict__ g, const float* __restrict__ bta,
    int total, int C, int HW, int relu)
{
  unsigned i = blockIdx.x * blockDim.x + threadIdx.x;
  if (i >= (unsigned)total) return;
  unsigned c = (i / (unsigned)HW) % (unsigned)C;
  float v = g[c] * (X[i] - mean[c]) * invstd[c] + bta[c];
  if (R) v += R[i];
  if (relu) v = fmaxf(v, 0.f);
  X[i] = v;
}

__global__ void maxpool3x3s2_kernel(
    const float* __restrict__ X, float* __restrict__ Y,
    int Bn, int C, int H, int W, int OH, int OW)
{
  unsigned i = blockIdx.x * blockDim.x + threadIdx.x;
  unsigned total = (unsigned)(Bn * C * OH * OW);
  if (i >= total) return;
  unsigned ow = i % (unsigned)OW; unsigned r = i / (unsigned)OW;
  unsigned oh = r % (unsigned)OH; r /= (unsigned)OH;
  unsigned c = r % (unsigned)C;   unsigned b = r / (unsigned)C;
  float m = -3.402823466e38f;
  for (int kh = 0; kh < 3; ++kh)
    for (int kw = 0; kw < 3; ++kw) {
      int ih = (int)(oh * 2) - 1 + kh, iw = (int)(ow * 2) - 1 + kw;
      if (ih >= 0 && ih < H && iw >= 0 && iw < W)
        m = fmaxf(m, X[((size_t)(b * C + c) * H + ih) * W + iw]);
    }
  Y[i] = m;
}

__global__ void concat_tokens_kernel(
    const float* __restrict__ t1, const float* __restrict__ t2,
    float* __restrict__ out, int Bn, int P, int E)
{
  unsigned i = blockIdx.x * blockDim.x + threadIdx.x;
  unsigned total = (unsigned)(Bn * 2 * P * E);
  if (i >= total) return;
  unsigned e = i % (unsigned)E; unsigned r = i / (unsigned)E;
  unsigned t = r % (unsigned)(2 * P); unsigned b = r / (unsigned)(2 * P);
  const float* src = (t < (unsigned)P) ? t1 : t2;
  unsigned tt = (t < (unsigned)P) ? t : t - P;
  out[i] = src[((size_t)b * E + e) * P + tt];
}

__global__ __launch_bounds__(256) void ln_residual_kernel(
    float* __restrict__ X, const float* __restrict__ R,
    const float* __restrict__ g, const float* __restrict__ bta, int E)
{
  __shared__ float red[256];
  int row = blockIdx.x, t = threadIdx.x;
  float* xr = X + (size_t)row * E;
  const float* rr = R + (size_t)row * E;
  float v0 = xr[t] + rr[t];
  float v1 = xr[t + 256] + rr[t + 256];
  red[t] = v0 + v1; __syncthreads();
  for (int o = 128; o > 0; o >>= 1) { if (t < o) red[t] += red[t + o]; __syncthreads(); }
  float mean = red[0] / E; __syncthreads();
  float d0 = v0 - mean, d1 = v1 - mean;
  red[t] = d0 * d0 + d1 * d1; __syncthreads();
  for (int o = 128; o > 0; o >>= 1) { if (t < o) red[t] += red[t + o]; __syncthreads(); }
  float inv = rsqrtf(red[0] / E + 1e-5f);
  xr[t] = g[t] * d0 * inv + bta[t];
  xr[t + 256] = g[t + 256] * d1 * inv + bta[t + 256];
}

__global__ __launch_bounds__(64) void attn_kernel(
    const float* __restrict__ QKV, float* __restrict__ O,
    int S, int N, int E, int NH)
{
  __shared__ float sc[64];
  __shared__ float pr[64];
  unsigned nh = blockIdx.x;
  unsigned n = nh / (unsigned)NH, h = nh % (unsigned)NH;
  int d = E / NH;
  int p = threadIdx.x;
  int s = p >> 3, t = p & 7;
  const float scale = rsqrtf((float)d);
  const float* qrow = QKV + ((size_t)(s * N + n)) * (3 * E) + h * d;
  const float* krow = QKV + ((size_t)(t * N + n)) * (3 * E) + E + h * d;
  float acc = 0.f;
  for (int e = 0; e < d; ++e) acc += qrow[e] * krow[e];
  sc[p] = acc * scale;
  __syncthreads();
  float mx = sc[s * 8];
  for (int k = 1; k < 8; ++k) mx = fmaxf(mx, sc[s * 8 + k]);
  float ex = expf(sc[p] - mx);
  pr[p] = ex; __syncthreads();
  float sum = 0.f;
  for (int k = 0; k < 8; ++k) sum += pr[s * 8 + k];
  float a = ex / sum;
  __syncthreads();
  pr[p] = a; __syncthreads();
  int s2 = p >> 3;
  for (int j = 0; j < 8; ++j) {
    int dd = (p & 7) + 8 * j;
    float o = 0.f;
    for (int tt = 0; tt < 8; ++tt)
      o += pr[s2 * 8 + tt] *
           QKV[((size_t)(tt * N + n)) * (3 * E) + 2 * E + h * d + dd];
    O[((size_t)(s2 * N + n)) * E + h * d + dd] = o;
  }
}

__global__ __launch_bounds__(128) void kmeans_select_kernel(
    const float* __restrict__ tok, float* __restrict__ cent_out, int T, int E)
{
  __shared__ float cent[4 * 512];
  __shared__ float sums[4 * 512];
  __shared__ float cnt[4];
  __shared__ float dall[128 * 4];
  __shared__ int bestid[4];
  int b = blockIdx.x, t = threadIdx.x;
  const float* tb = tok + (size_t)b * T * E;
  for (int i = t; i < 4 * E; i += 128) {
    int k = i / E, e = i % E;
    cent[i] = tb[(size_t)(k * (T / 4)) * E + e];
  }
  __syncthreads();
  for (int it = 0; it < 5; ++it) {
    float d[4] = {0.f, 0.f, 0.f, 0.f};
    for (int e = 0; e < E; ++e) {
      float x = tb[(size_t)t * E + e];
#pragma unroll
      for (int k = 0; k < 4; ++k) { float df = x - cent[k * E + e]; d[k] += df * df; }
    }
    int a = 0; float bd = d[0];
    for (int k = 1; k < 4; ++k) if (d[k] < bd) { bd = d[k]; a = k; }
    for (int i = t; i < 4 * E; i += 128) sums[i] = 0.f;
    if (t < 4) cnt[t] = 0.f;
    __syncthreads();
    for (int e = 0; e < E; ++e) atomicAdd(&sums[a * E + e], tb[(size_t)t * E + e]);
    atomicAdd(&cnt[a], 1.f);
    __syncthreads();
    for (int i = t; i < 4 * E; i += 128) {
      int k = i / E;
      if (cnt[k] > 0.f) cent[i] = sums[i] / fmaxf(cnt[k], 1.f);
    }
    __syncthreads();
  }
  {
    float d[4] = {0.f, 0.f, 0.f, 0.f};
    for (int e = 0; e < E; ++e) {
      float x = tb[(size_t)t * E + e];
#pragma unroll
      for (int k = 0; k < 4; ++k) { float df = x - cent[k * E + e]; d[k] += df * df; }
    }
    for (int k = 0; k < 4; ++k) dall[t * 4 + k] = d[k];
  }
  __syncthreads();
  if (t < 4) {
    float bd = dall[t]; int bi = 0;
    for (int tt = 1; tt < T; ++tt) {
      float v = dall[tt * 4 + t];
      if (v < bd) { bd = v; bi = tt; }
    }
    bestid[t] = bi;
  }
  __syncthreads();
  for (int i = t; i < 4 * E; i += 128) {
    int k = i / E, e = i % E;
    cent_out[((size_t)b * 4 + k) * E + e] = tb[(size_t)bestid[k] * E + e];
  }
}

__global__ __launch_bounds__(256) void classifier_loss_kernel(
    const float* __restrict__ cent, const float* __restrict__ Wc,
    const float* __restrict__ bc, const int* __restrict__ label,
    float* __restrict__ out, int Bn, int F, int NC)
{
  __shared__ float lg[80];
  __shared__ float lss[8];
  int t = threadIdx.x;
  if (t < Bn * NC) {
    int b = t / NC, c = t % NC;
    float a = bc[c];
    for (int e = 0; e < F; ++e)
      a += cent[(size_t)b * F + e] * Wc[(size_t)c * F + e];
    lg[t] = a; out[t] = a;
  }
  __syncthreads();
  if (t < Bn) {
    float mx = lg[t * NC];
    for (int c = 1; c < NC; ++c) mx = fmaxf(mx, lg[t * NC + c]);
    float se = 0.f;
    for (int c = 0; c < NC; ++c) se += expf(lg[t * NC + c] - mx);
    float lp = lg[t * NC + label[t]] - mx - logf(se);
    lss[t] = -lp;
  }
  __syncthreads();
  if (t == 0) {
    float s = 0.f;
    for (int b2 = 0; b2 < Bn; ++b2) s += lss[b2];
    out[Bn * NC] = s / Bn;
  }
}

// ============================================================================
extern "C" void kernel_launch(void* const* d_in, const int* in_sizes, int n_in,
                              void* d_out, int out_size, void* d_ws, size_t ws_size,
                              hipStream_t stream) {
  (void)in_sizes; (void)n_in; (void)out_size; (void)ws_size;

  auto P = [&](int i) -> const float* { return (const float*)d_in[i]; };
  const int* label = (const int*)d_in[2];
  const int CLS_B = 3, CLS_W = 4, ENC0 = 5, R1 = 41, R2 = 103;

  // ---- workspace bump allocator (requires ~296 MB) ----
  char* wsb = (char*)d_ws;
  size_t off = 0;
  auto alloc = [&](size_t bytes) -> void* {
    void* p = (void*)(wsb + off);
    off += (bytes + 255) & ~(size_t)255;
    return p;
  };
  float* big0 = (float*)alloc((size_t)8 * 64 * 256 * 256 * 4);   // conv1 raw
  float* bb[4];
  for (int i = 0; i < 4; ++i) bb[i] = (float*)alloc((size_t)8 * 64 * 128 * 128 * 4);
  float* statsM  = (float*)alloc(512 * 4);
  float* statsV  = (float*)alloc(512 * 4);
  float* tok1    = (float*)alloc((size_t)8 * 512 * 64 * 4);
  float* tok2    = (float*)alloc((size_t)8 * 512 * 64 * 4);
  float* xenc    = (float*)alloc((size_t)1024 * 512 * 4);
  float* qkvbuf  = (float*)alloc((size_t)1024 * 1536 * 4);
  float* attnbuf = (float*)alloc((size_t)1024 * 512 * 4);
  float* projbuf = (float*)alloc((size_t)1024 * 512 * 4);
  float* ffbuf   = (float*)alloc((size_t)1024 * 2048 * 4);
  float* centbuf = (float*)alloc((size_t)8 * 4 * 512 * 4);
  _Float16* whalf = (_Float16*)alloc((size_t)512 * 512 * 9 * 2);  // weight scratch

  auto toh = [&](const float* w, int n) -> const _Float16* {
    f32_to_f16_kernel<<<(n + 255) / 256, 256, 0, stream>>>(w, whalf, n);
    return whalf;
  };
  auto conv = [&](const float* X, const float* Wt, const float* bias, float* Y,
                  int Cin, int H, int W, int Cout, int KH, int KW, int s, int p) {
    int OH = (H + 2 * p - KH) / s + 1;
    int OW = (W + 2 * p - KW) / s + 1;
    int M = 8 * OH * OW;
    const _Float16* Wh = toh(Wt, Cout * Cin * KH * KW);
    dim3 grid((M + BM - 1) / BM, (Cout + BN - 1) / BN);
    conv_igemm_f16_wmma<<<grid, 256, 0, stream>>>(X, Wh, bias, Y, 8, Cin, H, W,
                                                  Cout, OH, OW, KH, KW, s, p);
  };
  auto bn = [&](float* X, const float* R, int beta_i, int gamma_i, int C, int HW,
                int relu) {
    bn_stats_kernel<<<C, 256, 0, stream>>>(X, statsM, statsV, 8, C, HW);
    int total = 8 * C * HW;
    bn_apply_kernel<<<(total + 255) / 256, 256, 0, stream>>>(
        X, R, statsM, statsV, P(gamma_i), P(beta_i), total, C, HW, relu);
  };
  auto gemm = [&](const float* A, const float* Wt, const float* bias, float* C,
                  int M, int N, int K, int relu) {
    const _Float16* Wh = toh(Wt, N * K);
    dim3 grid((M + BM - 1) / BM, (N + BN - 1) / BN);
    gemm_f16_wmma<<<grid, 256, 0, stream>>>(A, Wh, bias, C, M, N, K, relu);
  };

  auto runblock = [&](int curi, int base, int cin, int cout, int strd, int hin,
                      bool hasdown) -> int {
    int o[3], c2 = 0;
    for (int i = 0; i < 4; ++i) if (i != curi) o[c2++] = i;
    float* t0 = bb[o[0]]; float* t1 = bb[o[1]]; float* t2 = bb[o[2]];
    const float* cur = bb[curi];
    int hout = hin / strd;
    conv(cur, P(base + 4), nullptr, t0, cin, hin, hin, cout, 3, 3, strd, 1);
    bn(t0, nullptr, base + 0, base + 1, cout, hout * hout, 1);
    conv(t0, P(base + 5), nullptr, t1, cout, hout, hout, cout, 3, 3, 1, 1);
    const float* sc = cur;
    if (hasdown) {
      conv(cur, P(base + 6), nullptr, t2, cin, hin, hin, cout, 1, 1, strd, 0);
      bn(t2, nullptr, base + 7, base + 8, cout, hout * hout, 0);
      sc = t2;
    }
    bn(t1, sc, base + 2, base + 3, cout, hout * hout, 1);
    return o[1];
  };

  const int b0off[4] = {5, 17, 32, 47};
  const int b1off[4] = {11, 26, 41, 56};
  const int scin[4] = {64, 64, 128, 256};
  const int scout[4] = {64, 128, 256, 512};
  const int sstr[4] = {1, 2, 2, 2};
  const int shin[4] = {128, 128, 64, 32};

  auto run_branch = [&](const float* x, int r, float* tokens) {
    conv(x, P(r + 2), nullptr, big0, 3, 512, 512, 64, 7, 7, 2, 3);
    bn(big0, nullptr, r + 0, r + 1, 64, 256 * 256, 1);
    {
      int total = 8 * 64 * 128 * 128;
      maxpool3x3s2_kernel<<<(total + 255) / 256, 256, 0, stream>>>(
          big0, bb[0], 8, 64, 256, 256, 128, 128);
    }
    int curi = 0;
    for (int s = 0; s < 4; ++s) {
      curi = runblock(curi, r + b0off[s], scin[s], scout[s], sstr[s], shin[s], s > 0);
      curi = runblock(curi, r + b1off[s], scout[s], scout[s], 1, shin[s] / sstr[s], false);
    }
    conv(bb[curi], P(r + 4), P(r + 3), tokens, 512, 16, 16, 512, 2, 2, 2, 0);
  };

  run_branch(P(0), R1, tok1);
  run_branch(P(1), R2, tok2);

  {
    int total = 8 * 128 * 512;
    concat_tokens_kernel<<<(total + 255) / 256, 256, 0, stream>>>(
        tok1, tok2, xenc, 8, 64, 512);
  }

  for (int l = 0; l < 3; ++l) {
    int eb = ENC0 + 12 * l;
    gemm(xenc, P(eb + 11), P(eb + 10), qkvbuf, 1024, 1536, 512, 0);
    attn_kernel<<<128 * 8, 64, 0, stream>>>(qkvbuf, attnbuf, 8, 128, 512, 8);
    gemm(attnbuf, P(eb + 9), P(eb + 8), projbuf, 1024, 512, 512, 0);
    ln_residual_kernel<<<1024, 256, 0, stream>>>(xenc, projbuf, P(eb + 5), P(eb + 4), 512);
    gemm(xenc, P(eb + 1), P(eb + 0), ffbuf, 1024, 2048, 512, 1);
    gemm(ffbuf, P(eb + 3), P(eb + 2), projbuf, 1024, 512, 2048, 0);
    ln_residual_kernel<<<1024, 256, 0, stream>>>(xenc, projbuf, P(eb + 7), P(eb + 6), 512);
  }

  kmeans_select_kernel<<<8, 128, 0, stream>>>(xenc, centbuf, 128, 512);
  classifier_loss_kernel<<<1, 256, 0, stream>>>(
      centbuf, P(CLS_W), P(CLS_B), label, (float*)d_out, 8, 2048, 10);
}
